// DeepseekV2LiteExperts_13675175870987
// MI455X (gfx1250) — matrix operand order repaired and checked by reading, not compiled, not run on hardware
//
#include <hip/hip_runtime.h>
#include <stdint.h>

#define E_ 32
#define H_ 2048
#define I_ 1408
#define T_ 8192

#define BM 128
#define BN 128
#define BK 32
#define MAXMT (T_ / BM)

#if defined(__has_builtin)
#if __has_builtin(__builtin_amdgcn_global_load_async_to_lds_b128)
#define USE_ASYNC_LDS 1
#endif
#endif
#ifndef USE_ASYNC_LDS
#define USE_ASYNC_LDS 0
#endif

#define GLOBAL_AS __attribute__((address_space(1)))
#define LDS_AS    __attribute__((address_space(3)))

typedef __attribute__((ext_vector_type(16))) __bf16 v16bf;
typedef __attribute__((ext_vector_type(8)))  float  v8f;
typedef __attribute__((ext_vector_type(2)))  __bf16 bf16x2;
typedef __attribute__((ext_vector_type(4)))  int    v4i;

union FragAB {
  uint32_t u[8];
  uint4    q[2];
  v16bf    v;
};

// Hardware packed f32->bf16 (RNE): lowers to v_cvt_pk_bf16_f32
__device__ __forceinline__ uint32_t pack2(float a, float b) {
  bf16x2 t;
  t[0] = (__bf16)a;
  t[1] = (__bf16)b;
  return __builtin_bit_cast(uint32_t, t);
}

__device__ __forceinline__ void wait_async0() {
#if USE_ASYNC_LDS
#if __has_builtin(__builtin_amdgcn_s_wait_asynccnt)
  __builtin_amdgcn_s_wait_asynccnt(0);
#else
  asm volatile("s_wait_asynccnt 0x0" ::: "memory");
#endif
#endif
}

// ---------------------------------------------------------------------------
// Stage 1: h[t, i] = silu(x @ Wg[g]) * (x @ Wu[g])   (bf16 output to workspace)
// ---------------------------------------------------------------------------
__global__ __launch_bounds__(256) void moe_gate_up_kernel(
    const float* __restrict__ x, const float* __restrict__ Wg,
    const float* __restrict__ Wu, const int* __restrict__ offs,
    uint16_t* __restrict__ h) {
  const int g     = blockIdx.z;
  const int start = (g == 0) ? 0 : offs[g - 1];
  const int end   = offs[g];
  const int row0  = start + blockIdx.y * BM;
  if (row0 >= end) return;
  const int nbase = blockIdx.x * BN;

  __shared__ uint16_t As[2][BM * BK];   // tokens, bf16 [m][k]
  __shared__ uint16_t Bgs[2][BN * BK];  // gate weights, bf16 transposed [n][k]
  __shared__ uint16_t Bus[2][BN * BK];  // up   weights, bf16 transposed [n][k]

  const int t     = threadIdx.x;
  const int lane  = t & 31;
  const int wave  = t >> 5;
  const int wm    = wave & 3;   // 4 waves along M (32 rows each)
  const int wn    = wave >> 2;  // 2 waves along N (64 cols each)
  const int lhalf = lane >> 4;
  const int l16   = lane & 15;

  const size_t goff = (size_t)g * H_ * I_;

  float4 pa[4], pg[4], pu[4];  // register prefetch tiles

  auto loadA = [&](int k0) {
#pragma unroll
    for (int p = 0; p < 4; ++p) {
      const int row  = p * 32 + (t >> 3);
      const int k4   = (t & 7) * 4;
      const int grow = row0 + row;
      pa[p] = make_float4(0.f, 0.f, 0.f, 0.f);
      if (grow < end) pa[p] = *(const float4*)(x + (size_t)grow * H_ + k0 + k4);
    }
  };
  auto loadB = [&](int k0) {
#pragma unroll
    for (int u = 0; u < 2; ++u) {
      const int idx = t + 256 * u;
      const int k   = (idx >> 5) * 2;    // even K row
      const int n4  = (idx & 31) * 4;
      const size_t off = goff + (size_t)(k0 + k) * I_ + nbase + n4;
      pg[2 * u]     = *(const float4*)(Wg + off);
      pg[2 * u + 1] = *(const float4*)(Wg + off + I_);
      pu[2 * u]     = *(const float4*)(Wu + off);
      pu[2 * u + 1] = *(const float4*)(Wu + off + I_);
    }
  };
  auto stageStore = [&](int buf) {
#pragma unroll
    for (int p = 0; p < 4; ++p) {
      const int row = p * 32 + (t >> 3);
      const int k4  = (t & 7) * 4;
      *(uint2*)&As[buf][row * BK + k4] =
          make_uint2(pack2(pa[p].x, pa[p].y), pack2(pa[p].z, pa[p].w));
    }
#pragma unroll
    for (int u = 0; u < 2; ++u) {
      const int idx = t + 256 * u;
      const int k   = (idx >> 5) * 2;
      const int n4  = (idx & 31) * 4;
      const float4 g0 = pg[2 * u], g1 = pg[2 * u + 1];
      const float4 u0 = pu[2 * u], u1 = pu[2 * u + 1];
      *(uint32_t*)&Bgs[buf][(n4 + 0) * BK + k] = pack2(g0.x, g1.x);
      *(uint32_t*)&Bgs[buf][(n4 + 1) * BK + k] = pack2(g0.y, g1.y);
      *(uint32_t*)&Bgs[buf][(n4 + 2) * BK + k] = pack2(g0.z, g1.z);
      *(uint32_t*)&Bgs[buf][(n4 + 3) * BK + k] = pack2(g0.w, g1.w);
      *(uint32_t*)&Bus[buf][(n4 + 0) * BK + k] = pack2(u0.x, u1.x);
      *(uint32_t*)&Bus[buf][(n4 + 1) * BK + k] = pack2(u0.y, u1.y);
      *(uint32_t*)&Bus[buf][(n4 + 2) * BK + k] = pack2(u0.z, u1.z);
      *(uint32_t*)&Bus[buf][(n4 + 3) * BK + k] = pack2(u0.w, u1.w);
    }
  };

  v8f zero = {};
  v8f acc_g[2][4], acc_u[2][4];
#pragma unroll
  for (int i = 0; i < 2; ++i)
#pragma unroll
    for (int j = 0; j < 4; ++j) { acc_g[i][j] = zero; acc_u[i][j] = zero; }

  loadA(0);
  loadB(0);
  stageStore(0);

  const int NK = H_ / BK;
  for (int kt = 0; kt < NK; ++kt) {
    const int  cur  = kt & 1;
    const bool more = (kt + 1) < NK;
    if (more) { loadA((kt + 1) * BK); loadB((kt + 1) * BK); }  // in flight during compute
    __syncthreads();

    // Load ALL fragments into distinct registers: one clause of ds_load_b128,
    // one amortized dscnt wait, then an uninterrupted burst of 16 WMMAs.
    FragAB a[2], bg[4], bu[4];
#pragma unroll
    for (int ms = 0; ms < 2; ++ms) {
      const uint32_t* Ar = (const uint32_t*)&As[cur][(wm * 32 + ms * 16 + l16) * BK];
      a[ms].q[0] = *(const uint4*)&Ar[4 * lhalf];
      a[ms].q[1] = *(const uint4*)&Ar[8 + 4 * lhalf];
    }
#pragma unroll
    for (int ns = 0; ns < 4; ++ns) {
      const uint32_t* Bgr = (const uint32_t*)&Bgs[cur][(wn * 64 + ns * 16 + l16) * BK];
      const uint32_t* Bur = (const uint32_t*)&Bus[cur][(wn * 64 + ns * 16 + l16) * BK];
      bg[ns].q[0] = *(const uint4*)&Bgr[4 * lhalf];
      bg[ns].q[1] = *(const uint4*)&Bgr[8 + 4 * lhalf];
      bu[ns].q[0] = *(const uint4*)&Bur[4 * lhalf];
      bu[ns].q[1] = *(const uint4*)&Bur[8 + 4 * lhalf];
    }
#pragma unroll
    for (int ns = 0; ns < 4; ++ns)
#pragma unroll
      for (int ms = 0; ms < 2; ++ms) {
        acc_g[ms][ns] = __builtin_amdgcn_wmma_f32_16x16x32_bf16(
            false, a[ms].v, false, bg[ns].v, (short)0, acc_g[ms][ns], false, false);
        acc_u[ms][ns] = __builtin_amdgcn_wmma_f32_16x16x32_bf16(
            false, a[ms].v, false, bu[ns].v, (short)0, acc_u[ms][ns], false, false);
      }
    if (more) stageStore(cur ^ 1);  // safe: buffer nxt last read before this barrier
  }

  // ---- epilogue: silu(g) * u -> bf16 h
#pragma unroll
  for (int ms = 0; ms < 2; ++ms)
#pragma unroll
    for (int ns = 0; ns < 4; ++ns)
#pragma unroll
      for (int r = 0; r < 8; ++r) {
        const int row = row0 + wm * 32 + ms * 16 + r + 8 * lhalf;
        const int col = nbase + wn * 64 + ns * 16 + l16;
        if (row < end) {
          const float gv = acc_g[ms][ns][r];
          const float uv = acc_u[ms][ns][r];
          const float s  = gv / (1.0f + __expf(-gv));
          h[(size_t)row * I_ + col] = (uint16_t)(pack2(s * uv, 0.f) & 0xFFFFu);
        }
      }
}

// ---------------------------------------------------------------------------
// Stage 2: out[t, n] = h[t, :] @ Wd[g]   (f32 output)
// A tile is a pure bf16 copy -> use GLOBAL_LOAD_ASYNC_TO_LDS_B128 (ASYNCcnt)
// ---------------------------------------------------------------------------
__global__ __launch_bounds__(256) void moe_down_kernel(
    const uint16_t* __restrict__ h, const float* __restrict__ Wd,
    const int* __restrict__ offs, float* __restrict__ out) {
  const int g     = blockIdx.z;
  const int start = (g == 0) ? 0 : offs[g - 1];
  const int end   = offs[g];
  const int row0  = start + blockIdx.y * BM;
  if (row0 >= end) return;
  const int nbase = blockIdx.x * BN;

  __shared__ uint16_t As[2][BM * BK];
  __shared__ uint16_t Bs[2][BN * BK];

  const int t     = threadIdx.x;
  const int lane  = t & 31;
  const int wave  = t >> 5;
  const int wm    = wave & 3;
  const int wn    = wave >> 2;
  const int lhalf = lane >> 4;
  const int l16   = lane & 15;

  const size_t goff = (size_t)g * I_ * H_;

  float4 pb[4];

#if USE_ASYNC_LDS
  auto asyncLoadA = [&](int k0, int buf) {
#pragma unroll
    for (int p = 0; p < 2; ++p) {
      const int row  = p * 64 + (t >> 2);
      const int k8   = (t & 3) * 8;
      const int grow = row0 + row;
      if (grow < end) {  // EXEC-masked; invalid rows feed only discarded outputs
        __builtin_amdgcn_global_load_async_to_lds_b128(
            (GLOBAL_AS v4i*)(void*)(h + (size_t)grow * I_ + k0 + k8),
            (LDS_AS v4i*)(void*)&As[buf][row * BK + k8], 0, 0);
      }
    }
  };
#else
  uint4 pa[2];
  auto loadA = [&](int k0) {
#pragma unroll
    for (int p = 0; p < 2; ++p) {
      const int row  = p * 64 + (t >> 2);
      const int k8   = (t & 3) * 8;
      const int grow = row0 + row;
      pa[p] = make_uint4(0u, 0u, 0u, 0u);
      if (grow < end) pa[p] = *(const uint4*)(h + (size_t)grow * I_ + k0 + k8);
    }
  };
  auto storeA = [&](int buf) {
#pragma unroll
    for (int p = 0; p < 2; ++p) {
      const int row = p * 64 + (t >> 2);
      const int k8  = (t & 3) * 8;
      *(uint4*)&As[buf][row * BK + k8] = pa[p];
    }
  };
#endif

  auto loadB = [&](int k0) {
#pragma unroll
    for (int u = 0; u < 2; ++u) {
      const int idx = t + 256 * u;
      const int k   = (idx >> 5) * 2;
      const int n4  = (idx & 31) * 4;
      const size_t off = goff + (size_t)(k0 + k) * H_ + nbase + n4;
      pb[2 * u]     = *(const float4*)(Wd + off);
      pb[2 * u + 1] = *(const float4*)(Wd + off + H_);
    }
  };
  auto storeB = [&](int buf) {
#pragma unroll
    for (int u = 0; u < 2; ++u) {
      const int idx = t + 256 * u;
      const int k   = (idx >> 5) * 2;
      const int n4  = (idx & 31) * 4;
      const float4 w0 = pb[2 * u], w1 = pb[2 * u + 1];
      *(uint32_t*)&Bs[buf][(n4 + 0) * BK + k] = pack2(w0.x, w1.x);
      *(uint32_t*)&Bs[buf][(n4 + 1) * BK + k] = pack2(w0.y, w1.y);
      *(uint32_t*)&Bs[buf][(n4 + 2) * BK + k] = pack2(w0.z, w1.z);
      *(uint32_t*)&Bs[buf][(n4 + 3) * BK + k] = pack2(w0.w, w1.w);
    }
  };

  v8f zero = {};
  v8f acc[2][4];
#pragma unroll
  for (int i = 0; i < 2; ++i)
#pragma unroll
    for (int j = 0; j < 4; ++j) acc[i][j] = zero;

#if USE_ASYNC_LDS
  asyncLoadA(0, 0);
#else
  loadA(0);
#endif
  loadB(0);
#if !USE_ASYNC_LDS
  storeA(0);
#endif
  storeB(0);

  const int NK = I_ / BK;
  for (int kt = 0; kt < NK; ++kt) {
    const int  cur  = kt & 1;
    const bool more = (kt + 1) < NK;
    if (more) {
      loadB((kt + 1) * BK);
#if !USE_ASYNC_LDS
      loadA((kt + 1) * BK);
#endif
    }
    wait_async0();       // drain our async writes before making them visible
    __syncthreads();
#if USE_ASYNC_LDS
    // Issue next A tile only AFTER the barrier: other waves may read buf nxt
    // up until they arrive. It lands while we run this iteration's WMMAs.
    if (more) asyncLoadA((kt + 1) * BK, cur ^ 1);
#endif

    FragAB a[2], b[4];
#pragma unroll
    for (int ms = 0; ms < 2; ++ms) {
      const uint32_t* Ar = (const uint32_t*)&As[cur][(wm * 32 + ms * 16 + l16) * BK];
      a[ms].q[0] = *(const uint4*)&Ar[4 * lhalf];
      a[ms].q[1] = *(const uint4*)&Ar[8 + 4 * lhalf];
    }
#pragma unroll
    for (int ns = 0; ns < 4; ++ns) {
      const uint32_t* Br = (const uint32_t*)&Bs[cur][(wn * 64 + ns * 16 + l16) * BK];
      b[ns].q[0] = *(const uint4*)&Br[4 * lhalf];
      b[ns].q[1] = *(const uint4*)&Br[8 + 4 * lhalf];
    }
#pragma unroll
    for (int ns = 0; ns < 4; ++ns)
#pragma unroll
      for (int ms = 0; ms < 2; ++ms) {
        acc[ms][ns] = __builtin_amdgcn_wmma_f32_16x16x32_bf16(
            false, a[ms].v, false, b[ns].v, (short)0, acc[ms][ns], false, false);
      }
    if (more) {
#if !USE_ASYNC_LDS
      storeA(cur ^ 1);
#endif
      storeB(cur ^ 1);
    }
  }

#pragma unroll
  for (int ms = 0; ms < 2; ++ms)
#pragma unroll
    for (int ns = 0; ns < 4; ++ns)
#pragma unroll
      for (int r = 0; r < 8; ++r) {
        const int row = row0 + wm * 32 + ms * 16 + r + 8 * lhalf;
        const int col = nbase + wn * 64 + ns * 16 + l16;
        if (row < end) out[(size_t)row * H_ + col] = acc[ms][ns][r];
      }
}

extern "C" void kernel_launch(void* const* d_in, const int* in_sizes, int n_in,
                              void* d_out, int out_size, void* d_ws, size_t ws_size,
                              hipStream_t stream) {
  (void)in_sizes; (void)n_in; (void)out_size; (void)ws_size;
  const float* x    = (const float*)d_in[0];
  const float* Wg   = (const float*)d_in[1];
  const float* Wu   = (const float*)d_in[2];
  const float* Wd   = (const float*)d_in[3];
  const int*   offs = (const int*)d_in[4];
  float*       out  = (float*)d_out;
  uint16_t*    h    = (uint16_t*)d_ws;  // T_*I_ bf16 = 23 MB intermediate

  dim3 block(256);
  dim3 grid1(I_ / BN, MAXMT, E_);
  dim3 grid2(H_ / BN, MAXMT, E_);
  moe_gate_up_kernel<<<grid1, block, 0, stream>>>(x, Wg, Wu, offs, h);
  moe_down_kernel<<<grid2, block, 0, stream>>>(h, Wd, offs, out);
}
